// LSTMKF_PhysicsVT_68659347194182
// MI455X (gfx1250) — matrix operand order repaired and checked by reading, not compiled, not run on hardware
//
#include <hip/hip_runtime.h>
#include <hip/hip_bf16.h>
#include <math.h>

typedef __attribute__((ext_vector_type(16))) _Float16 v16h;
typedef __attribute__((ext_vector_type(8)))  _Float16 v8h;
typedef __attribute__((ext_vector_type(8)))  float    v8f;

static constexpr int kB   = 2048;
static constexpr int kHH  = 30;
static constexpr int kT   = 50;
static constexpr int kD   = 14;
static constexpr int kHID = 256;
static constexpr int kK0  = 288;   // 256 (h) + 14 (x) + 18 pad, multiple of 32
static constexpr int kK1  = 512;   // 256 (h) + 256 (layer0 h)
static constexpr float kTS   = 0.1f;
static constexpr float kJIT  = 0.001f;
static constexpr float kMINQ = 1e-5f;
static constexpr float kMINR = 1e-4f;

__device__ __forceinline__ float sigm(float x) { return 1.0f / (1.0f + __expf(-x)); }

__device__ __forceinline__ v8f wmma16(v16h a, v16h b, v8f c) {
  // emits v_wmma_f32_16x16x32_f16
  return __builtin_amdgcn_wmma_f32_16x16x32_f16(false, a, false, b, (short)0, c,
                                                false, false);
}

// A-operand fragment (16x32 f16, M=batch, K): lane holds row M=lane%16;
// halves live at K offsets {base..base+7, base+8..base+15}, base = 16*(lane/16).
// Two ds_load_b128 per fragment.
template<int KCAT>
__device__ __forceinline__ v16h load_a_frag(const _Float16* Ab, int k0, int lr, int hi) {
  const _Float16* p = Ab + lr * KCAT + k0 + 16 * hi;
  union { v16h v; v8h h[2]; } u;
  u.h[0] = *(const v8h*)(p);
  u.h[1] = *(const v8h*)(p + 8);
  return u.v;
}

// B-operand fragment (32x16 f16, K x N): lane holds column N=lane%16 with 16
// consecutive K values starting at k0 + 16*(lane/16). Since B[k][n] = W[n][k]
// (W row-major [1024][KCAT]), this is one contiguous 32-byte region per lane
// -> two global_load_b128.
__device__ __forceinline__ v16h load_b16(const _Float16* p) {
  union { v16h v; v8h h[2]; } u;
  u.h[0] = *(const v8h*)(p);
  u.h[1] = *(const v8h*)(p + 8);
  return u.v;
}

// One LSTM layer step for a 16-row batch tile. Each wave owns hidden tiles
// {w, w+8}; for each it accumulates the 4 gate tiles (i,f,g,o) so the gate
// nonlinearity and cell state stay entirely in C/D-fragment registers.
template<int KCAT>
__device__ __forceinline__ void lstm_step(const _Float16* Ab, const _Float16* Wc,
                                          const float* __restrict__ bias,
                                          v8f (&cst)[2],
                                          _Float16* hOwn, int ownK,
                                          _Float16* hNext, int nextK, int nextOff,
                                          float* hlast_out,
                                          int b0, int w, int lr, int hi) {
  v8f acc[2][4];
  v8f zf = {};
#pragma unroll
  for (int hti = 0; hti < 2; ++hti)
#pragma unroll
    for (int g = 0; g < 4; ++g) acc[hti][g] = zf;

#pragma unroll
  for (int hti = 0; hti < 2; ++hti) {
    const int ht = w + 8 * hti;
    // Per-gate base pointers: row (gate*256 + ht*16 + lr) of W, this lane's
    // K-half starts at 16*hi. Stepped by k0 inside the loop.
    const _Float16* bp0 = Wc + (size_t)(0 * 256 + ht * 16 + lr) * KCAT + 16 * hi;
    const _Float16* bp1 = Wc + (size_t)(1 * 256 + ht * 16 + lr) * KCAT + 16 * hi;
    const _Float16* bp2 = Wc + (size_t)(2 * 256 + ht * 16 + lr) * KCAT + 16 * hi;
    const _Float16* bp3 = Wc + (size_t)(3 * 256 + ht * 16 + lr) * KCAT + 16 * hi;
    // Keep this loop rolled: only one chunk's A (2x ds_b128) and B
    // (8x global_b128) operands in flight -> no spills to scratch.
#pragma clang loop unroll(disable)
    for (int k0 = 0; k0 < KCAT; k0 += 32) {
      v16h a = load_a_frag<KCAT>(Ab, k0, lr, hi);
      if (k0 + 32 < KCAT)
        // locality=3 -> WGP-scope prefetch: pulls next chunk of gate-0
        // weights into all cache levels (SYS/SE-scope would stop at GL2,
        // where the packed weights already reside).
        __builtin_prefetch(bp0 + k0 + 32, 0, 3);
      acc[hti][0] = wmma16(a, load_b16(bp0 + k0), acc[hti][0]);
      acc[hti][1] = wmma16(a, load_b16(bp1 + k0), acc[hti][1]);
      acc[hti][2] = wmma16(a, load_b16(bp2 + k0), acc[hti][2]);
      acc[hti][3] = wmma16(a, load_b16(bp3 + k0), acc[hti][3]);
    }
  }
  __syncthreads();   // all waves finished reading Ab before h is overwritten

#pragma unroll
  for (int hti = 0; hti < 2; ++hti) {
    const int ht = w + 8 * hti;
    const int nc = ht * 16 + lr;
    const float bi = bias[nc],        bfv = bias[256 + nc];
    const float bg = bias[512 + nc],  bo  = bias[768 + nc];
#pragma unroll
    for (int v = 0; v < 8; ++v) {
      float iv = sigm(acc[hti][0][v] + bi);
      float fv = sigm(acc[hti][1][v] + bfv);
      float gv = tanhf(acc[hti][2][v] + bg);
      float ov = sigm(acc[hti][3][v] + bo);
      float cc = fv * cst[hti][v] + iv * gv;
      cst[hti][v] = cc;
      float hv = ov * tanhf(cc);
      const int M = v + 8 * hi;              // C/D layout: M = vgpr + 8*(lane>=16)
      _Float16 hh = (_Float16)hv;
      hOwn[M * ownK + nc] = hh;              // recurrent input for next step
      if (hNext) hNext[M * nextK + nextOff + nc] = hh;   // feed layer 1
      if (hlast_out) hlast_out[(size_t)(b0 + M) * kHID + nc] = hv;
    }
  }
  __syncthreads();
}

// ---------------- weight packing ----------------
__global__ void pack_w0_kernel(const float* __restrict__ Wih, const float* __restrict__ Whh,
                               _Float16* __restrict__ Wcat) {
  int i = blockIdx.x * blockDim.x + threadIdx.x;
  if (i >= 1024 * kK0) return;
  int j = i / kK0, k = i % kK0;
  float v = (k < 256) ? Whh[j * 256 + k] : ((k < 270) ? Wih[j * kD + (k - 256)] : 0.0f);
  Wcat[i] = (_Float16)v;
}

__global__ void pack_w1_kernel(const float* __restrict__ Wih, const float* __restrict__ Whh,
                               _Float16* __restrict__ Wcat) {
  int i = blockIdx.x * blockDim.x + threadIdx.x;
  if (i >= 1024 * kK1) return;
  int j = i / kK1, k = i % kK1;
  float v = (k < 256) ? Whh[j * 256 + k] : Wih[j * 256 + (k - 256)];
  Wcat[i] = (_Float16)v;
}

__global__ void pack_bias_kernel(const float* b0i, const float* b0h,
                                 const float* b1i, const float* b1h,
                                 float* bias0, float* bias1) {
  int i = blockIdx.x * blockDim.x + threadIdx.x;
  if (i < 1024) bias0[i] = b0i[i] + b0h[i];
  else if (i < 2048) { int j = i - 1024; bias1[j] = b1i[j] + b1h[j]; }
}

__global__ void mask_kernel(const int* __restrict__ zm, float* __restrict__ mask14) {
  int i = blockIdx.x * blockDim.x + threadIdx.x;   // over B*T*7
  if (i >= kB * kT * 7) return;
  int k = i % 7, bt = i / 7;
  float v = (float)zm[bt * 21 + 3 * k + 2];
  mask14[bt * kD + 2 * k] = v;
  mask14[bt * kD + 2 * k + 1] = v;
}

// ---------------- fused 2-layer LSTM scan ----------------
__global__ __launch_bounds__(256)
void lstm_fused_kernel(const float* __restrict__ x_hist,
                       const _Float16* __restrict__ Wcat0,
                       const _Float16* __restrict__ Wcat1,
                       const float* __restrict__ bias0,
                       const float* __restrict__ bias1,
                       float* __restrict__ h_last) {
  __shared__ _Float16 A0[16 * kK0];   // [h(256) | x(14) | pad]
  __shared__ _Float16 A1[16 * kK1];   // [h1(256) | h0(256)]
  const int b0 = blockIdx.x * 16;
  const int tid = threadIdx.x;
  const int lane = tid & 31, w = tid >> 5;
  const int lr = lane & 15, hi = lane >> 4;

  for (int i = tid; i < 16 * kK0; i += 256) A0[i] = (_Float16)0.0f;
  for (int i = tid; i < 16 * kK1; i += 256) A1[i] = (_Float16)0.0f;

  v8f zf = {};
  v8f c0[2] = {zf, zf};
  v8f c1[2] = {zf, zf};
  __syncthreads();

  for (int t = 0; t < kHH; ++t) {
    if (tid < 16 * kD) {
      int row = tid / kD, col = tid % kD;
      A0[row * kK0 + 256 + col] =
          (_Float16)x_hist[((size_t)(b0 + row) * kHH + t) * kD + col];
    }
    __syncthreads();
    lstm_step<kK0>(A0, Wcat0, bias0, c0, A0, kK0, A1, kK1, 256,
                   nullptr, b0, w, lr, hi);
    lstm_step<kK1>(A1, Wcat1, bias1, c1, A1, kK1, nullptr, 0, 0,
                   (t == kHH - 1) ? h_last : nullptr, b0, w, lr, hi);
  }
}

// ---------------- policy head ----------------
__global__ void policy_kernel(const float* __restrict__ h_last,
                              const float* __restrict__ Wp,
                              const float* __restrict__ bp,
                              float* __restrict__ pol) {
  int i = blockIdx.x * blockDim.x + threadIdx.x;
  if (i >= kB * 6) return;
  int b = i / 6, k = i % 6;
  const float* h = h_last + (size_t)b * kHID;
  const float* wv = Wp + k * kHID;
  float s = bp[k];
  for (int j = 0; j < kHID; ++j) s += h[j] * wv[j];
  pol[i] = s;
}

// ---------------- physics rollout -> x_prior ----------------
__global__ void rollout_kernel(const float* __restrict__ x_hist,
                               const float* __restrict__ pol,
                               const float* __restrict__ centers,
                               float* __restrict__ x_prior) {
  int b = blockIdx.x * blockDim.x + threadIdx.x;
  if (b >= kB) return;
  const float* pb = pol + b * 6;
  const float K0 = pb[0], K1 = pb[1];
  const float L0 = pb[2], L1 = pb[3], L2 = pb[4], vref = pb[5];
  const float* xh = x_hist + (size_t)b * kHH * kD;
  const float x_last = xh[29 * kD + 0];
  const float y_last = xh[29 * kD + 1];

  float best = fabsf(x_last - centers[0]); int bi = 0;
  for (int c = 1; c < 3; ++c) { float d = fabsf(x_last - centers[c]); if (d < best) { best = d; bi = c; } }
  const float pref = centers[bi];

  const float Am[9] = {1.f, kTS, 0.5f * kTS * kTS, 0.f, 1.f, kTS, 0.f, 0.f, 1.f};
  const float Bm[3] = {0.f, 0.5f * kTS * kTS, kTS};
  const float kv[3] = {0.f, K0, K1};
  const float kl[3] = {L0, L1, L2};
  float Flon[9], Flat[9], elon[3], elat[3];
  for (int i = 0; i < 3; ++i) {
    for (int j = 0; j < 3; ++j) {
      Flon[i * 3 + j] = Am[i * 3 + j] - Bm[i] * kv[j];
      Flat[i * 3 + j] = Am[i * 3 + j] - Bm[i] * kl[j];
    }
    elon[i] = Bm[i] * (K0 * vref);
    elat[i] = Bm[i] * (L0 * pref);
  }
  float xl[3] = {x_last, 0.f, 0.f};
  float xt[3] = {y_last, 0.f, 0.f};
  float last[kD], vel[kD];
  for (int d = 0; d < kD; ++d) {
    last[d] = xh[29 * kD + d];
    vel[d] = last[d] - xh[28 * kD + d];
  }
  for (int t = 0; t < kT; ++t) {
    float nl[3], nt[3];
    for (int i = 0; i < 3; ++i) {
      nl[i] = Flon[i*3]*xl[0] + Flon[i*3+1]*xl[1] + Flon[i*3+2]*xl[2] + elon[i];
      nt[i] = Flat[i*3]*xt[0] + Flat[i*3+1]*xt[1] + Flat[i*3+2]*xt[2] + elat[i];
    }
    for (int i = 0; i < 3; ++i) { xl[i] = nl[i]; xt[i] = nt[i]; }
    float* op = x_prior + ((size_t)b * kT + t) * kD;
    for (int d = 0; d < kD; ++d) op[d] = last[d] + vel[d] * (float)(t + 1);
    op[0] = xl[0];
    op[1] = xt[0];
  }
}

// ---------------- q/r MLP: softplus(relu(x W1^T + b1) W2^T + b2), clamped ----
__global__ void mlp_kernel(const float* __restrict__ x,
                           const float* __restrict__ W1, const float* __restrict__ b1,
                           const float* __restrict__ W2, const float* __restrict__ b2,
                           float* __restrict__ out, float minval, int nrows) {
  int r = blockIdx.x * blockDim.x + threadIdx.x;
  if (r >= nrows) return;
  float xin[kD];
  for (int k = 0; k < kD; ++k) xin[k] = x[(size_t)r * kD + k];
  float h[64];
  for (int j = 0; j < 64; ++j) {
    float s = b1[j];
    for (int k = 0; k < kD; ++k) s += W1[j * kD + k] * xin[k];
    h[j] = fmaxf(s, 0.0f);
  }
  for (int o = 0; o < kD; ++o) {
    float s = b2[o];
    for (int j = 0; j < 64; ++j) s += W2[o * 64 + j] * h[j];
    float sp = fmaxf(s, 0.0f) + log1pf(__expf(-fabsf(s)));   // stable softplus
    out[(size_t)r * kD + o] = fmaxf(sp, minval);
  }
}

// ---------------- Kalman filter scan (per-batch, carried P) ----------------
__global__ void kf_kernel(const float* __restrict__ x_prior,
                          const float* __restrict__ q_diag,
                          const float* __restrict__ z_seq,
                          const float* __restrict__ r_diag,
                          const float* __restrict__ mask14,
                          float* __restrict__ x_post,
                          float* __restrict__ P_post) {
  int b = blockIdx.x * blockDim.x + threadIdx.x;
  if (b >= kB) return;
  float P[kD][kD];
  for (int i = 0; i < kD; ++i)
    for (int j = 0; j < kD; ++j) P[i][j] = (i == j) ? 1.0f : 0.0f;

  for (int t = 0; t < kT; ++t) {
    const size_t bt = (size_t)b * kT + t;
    const float* q  = q_diag + bt * kD;
    const float* z  = z_seq  + bt * kD;
    const float* rr = r_diag + bt * kD;
    const float* m  = mask14 + bt * kD;
    const float* xp = x_prior + bt * kD;

    float Ps[kD][kD];
    for (int i = 0; i < kD; ++i)
      for (int j = 0; j < kD; ++j)
        Ps[i][j] = 0.5f * (P[i][j] + P[j][i]) + ((i == j) ? q[i] : 0.0f);

    float Sreg[kD][kD];
    float tr = 0.0f;
    for (int i = 0; i < kD; ++i) tr += m[i] * (Ps[i][i] + rr[i] + kJIT);
    tr = fmaxf(tr, 1e-6f);
    for (int i = 0; i < kD; ++i)
      for (int j = 0; j < kD; ++j) {
        float s = m[i] * m[j] * Ps[i][j];
        if (i == j) s += m[i] * (rr[i] + kJIT) + (1.0f - m[i]) + kJIT * tr * m[i];
        Sreg[i][j] = s;
      }

    // Cholesky (lower) of Sreg into L
    float L[kD][kD];
    for (int i = 0; i < kD; ++i)
      for (int j = 0; j < kD; ++j) L[i][j] = Sreg[i][j];
    for (int k = 0; k < kD; ++k) {
      float d = sqrtf(fmaxf(L[k][k], 1e-30f));
      L[k][k] = d;
      float inv = 1.0f / d;
      for (int i = k + 1; i < kD; ++i) L[i][k] *= inv;
      for (int j = k + 1; j < kD; ++j)
        for (int i = j; i < kD; ++i) L[i][j] -= L[i][k] * L[j][k];
    }

    // K = (cho_solve(L, Pm^T))^T, Pm^T[i][j] = Ps[i][j]*m[i]
    float K[kD][kD];
    for (int j = 0; j < kD; ++j) {
      float y[kD];
      for (int i = 0; i < kD; ++i) {
        float s = Ps[i][j] * m[i];
        for (int k = 0; k < i; ++k) s -= L[i][k] * y[k];
        y[i] = s / L[i][i];
      }
      for (int i = kD - 1; i >= 0; --i) {
        float s = y[i];
        for (int k = i + 1; k < kD; ++k) s -= L[k][i] * y[k];
        y[i] = s / L[i][i];
      }
      for (int i = 0; i < kD; ++i) K[j][i] = y[i];   // K = X^T
    }

    float innov[kD];
    for (int i = 0; i < kD; ++i) innov[i] = m[i] * (z[i] - xp[i]);
    float* xo = x_post + bt * kD;
    for (int i = 0; i < kD; ++i) {
      float s = xp[i];
      for (int j = 0; j < kD; ++j) s += K[i][j] * innov[j];
      xo[i] = s;
    }

    float KS[kD][kD];
    for (int i = 0; i < kD; ++i)
      for (int j = 0; j < kD; ++j) {
        float s = 0.0f;
        for (int k = 0; k < kD; ++k) s += K[i][k] * Sreg[k][j];
        KS[i][j] = s;
      }
    float Pp[kD][kD];
    for (int i = 0; i < kD; ++i)
      for (int j = 0; j < kD; ++j) {
        float s = Ps[i][j];
        for (int k = 0; k < kD; ++k) s -= KS[i][k] * K[j][k];
        Pp[i][j] = s;
      }
    float* Po = P_post + bt * kD * kD;
    for (int i = 0; i < kD; ++i)
      for (int j = 0; j < kD; ++j) {
        float v = 0.5f * (Pp[i][j] + Pp[j][i]);
        P[i][j] = v;
        Po[i * kD + j] = v;
      }
  }
}

extern "C" void kernel_launch(void* const* d_in, const int* in_sizes, int n_in,
                              void* d_out, int out_size, void* d_ws, size_t ws_size,
                              hipStream_t stream) {
  (void)in_sizes; (void)n_in; (void)out_size; (void)ws_size;
  const float* x_hist = (const float*)d_in[0];
  const float* z_seq  = (const float*)d_in[1];
  const int*   zmask  = (const int*)d_in[2];
  const float* lanec  = (const float*)d_in[3];
  const float* Wih0 = (const float*)d_in[4];
  const float* Whh0 = (const float*)d_in[5];
  const float* bih0 = (const float*)d_in[6];
  const float* bhh0 = (const float*)d_in[7];
  const float* Wih1 = (const float*)d_in[8];
  const float* Whh1 = (const float*)d_in[9];
  const float* bih1 = (const float*)d_in[10];
  const float* bhh1 = (const float*)d_in[11];
  const float* Wp = (const float*)d_in[12];
  const float* bp = (const float*)d_in[13];
  const float* qW1 = (const float*)d_in[14];
  const float* qb1 = (const float*)d_in[15];
  const float* qW2 = (const float*)d_in[16];
  const float* qb2 = (const float*)d_in[17];
  const float* rW1 = (const float*)d_in[18];
  const float* rb1 = (const float*)d_in[19];
  const float* rW2 = (const float*)d_in[20];
  const float* rb2 = (const float*)d_in[21];

  // workspace layout
  _Float16* Wcat0 = (_Float16*)d_ws;                 // 1024*288
  _Float16* Wcat1 = Wcat0 + 1024 * kK0;              // 1024*512
  float* bias0  = (float*)(Wcat1 + 1024 * kK1);
  float* bias1  = bias0 + 1024;
  float* h_last = bias1 + 1024;                      // 2048*256
  float* pol    = h_last + (size_t)kB * kHID;        // 2048*6
  float* mask14 = pol + kB * 6;                      // 2048*50*14

  // output layout (reference return order)
  float* out = (float*)d_out;
  const size_t nBTD = (size_t)kB * kT * kD;          // 1,433,600
  float* x_prior = out;
  float* x_post  = out + nBTD;
  float* P_post  = out + 2 * nBTD;
  float* q_diag  = out + 2 * nBTD + (size_t)kB * kT * kD * kD;
  float* r_diag  = q_diag + nBTD;

  // 1) pack weights/biases to f16 concatenated layout
  pack_w0_kernel<<<(1024 * kK0 + 255) / 256, 256, 0, stream>>>(Wih0, Whh0, Wcat0);
  pack_w1_kernel<<<(1024 * kK1 + 255) / 256, 256, 0, stream>>>(Wih1, Whh1, Wcat1);
  pack_bias_kernel<<<(2048 + 255) / 256, 256, 0, stream>>>(bih0, bhh0, bih1, bhh1,
                                                           bias0, bias1);
  // 2) mask14 from z_mask_full21
  mask_kernel<<<(kB * kT * 7 + 255) / 256, 256, 0, stream>>>(zmask, mask14);

  // 3) fused 2-layer LSTM scan (WMMA f16 -> f32)
  lstm_fused_kernel<<<kB / 16, 256, 0, stream>>>(x_hist, Wcat0, Wcat1, bias0, bias1,
                                                 h_last);
  // 4) policy head
  policy_kernel<<<(kB * 6 + 255) / 256, 256, 0, stream>>>(h_last, Wp, bp, pol);

  // 5) physics rollout -> x_prior
  rollout_kernel<<<(kB + 63) / 64, 64, 0, stream>>>(x_hist, pol, lanec, x_prior);

  // 6) q / r MLPs
  const int nrows = kB * kT;
  mlp_kernel<<<(nrows + 255) / 256, 256, 0, stream>>>(x_prior, qW1, qb1, qW2, qb2,
                                                      q_diag, kMINQ, nrows);
  mlp_kernel<<<(nrows + 255) / 256, 256, 0, stream>>>(z_seq, rW1, rb1, rW2, rb2,
                                                      r_diag, kMINR, nrows);

  // 7) Kalman scan -> x_post, P_post
  kf_kernel<<<(kB + 63) / 64, 64, 0, stream>>>(x_prior, q_diag, z_seq, r_diag,
                                               mask14, x_post, P_post);
}
